// SoftGate_12618613916177
// MI455X (gfx1250) — matrix operand order repaired and checked by reading, not compiled, not run on hardware
//
#include <hip/hip_runtime.h>
#include <hip/hip_bf16.h>

typedef __attribute__((ext_vector_type(16))) _Float16 v16h;
typedef __attribute__((ext_vector_type(8)))  _Float16 v8h;
typedef __attribute__((ext_vector_type(8)))  float    v8f;
typedef __attribute__((ext_vector_type(4)))  float    v4f;

#define BB 8
#define SS 4096
#define DD 1024
#define DH 512
#define NTOK (BB*SS)   // 32768

// ---------------------------------------------------------------------------
// Kernel 1: pack W1 (f32 [D][Dh] row-major) into W1T (f16 [Dh][D] row-major)
// so WMMA B-fragments are per-lane contiguous. Also zero the ys accumulator.
// ---------------------------------------------------------------------------
__global__ __launch_bounds__(256)
void pack_w1(const float* __restrict__ w1, _Float16* __restrict__ w1t,
             float* __restrict__ ys) {
    int i = blockIdx.x * 256 + threadIdx.x;      // over DH*DD = 524288
    int n = i >> 10;                             // / DD
    int k = i & (DD - 1);
    w1t[(long)n * DD + k] = (_Float16)w1[(long)k * DH + n];
    if (i < NTOK) ys[i] = 0.0f;
}

// ---------------------------------------------------------------------------
// Kernel 2: fused gate-MLP. One wave computes a 16-token x 128-N strip of
// h = relu(x*W1 + b1) entirely in WMMA accumulators, then reduces against W2
// in-register and atomically accumulates the per-token gate logit into ys.
// ---------------------------------------------------------------------------
__global__ __launch_bounds__(128)
void gate_mlp_wmma(const float* __restrict__ x,        // [NTOK, D] f32
                   const _Float16* __restrict__ w1t,   // [DH, D]   f16
                   const float* __restrict__ b1,       // [DH]
                   const float* __restrict__ w2,       // [DH]
                   float* __restrict__ ys)             // [NTOK] accumulator
{
    const int lane  = threadIdx.x & 31;
    const int wave  = blockIdx.x * 4 + (threadIdx.x >> 5);
    const int strip = wave & 3;          // which 128-wide N strip
    const int mtile = wave >> 2;         // which 16-token tile
    const long t0   = (long)mtile * 16;
    const int  n0   = strip * 128;

    const int mrow  = lane & 15;         // A-matrix row held by this lane
    const int khalf = lane >> 4;         // A: K-half select; B: same K-half select

    v8f acc[8];
#pragma unroll
    for (int t = 0; t < 8; ++t) acc[t] = (v8f)0.0f;

    const float* xrow = x + (t0 + mrow) * DD;

    for (int kk = 0; kk < DD; kk += 32) {
        // A fragment (16x32 f16): lane holds K = kk+khalf*8+{0..7} and +16.
        const int ka = kk + khalf * 8;
        v4f a0 = *(const v4f*)(xrow + ka);
        v4f a1 = *(const v4f*)(xrow + ka + 4);
        v4f a2 = *(const v4f*)(xrow + ka + 16);
        v4f a3 = *(const v4f*)(xrow + ka + 20);
        v16h afrag;
#pragma unroll
        for (int i = 0; i < 4; ++i) {
            afrag[i]      = (_Float16)a0[i];
            afrag[4 + i]  = (_Float16)a1[i];
            afrag[8 + i]  = (_Float16)a2[i];
            afrag[12 + i] = (_Float16)a3[i];
        }
#pragma unroll
        for (int t = 0; t < 8; ++t) {
            // B fragment (32x16 f16): lane holds K = kk+khalf*8+{0..7} and +16 of column n.
            const int n = n0 + t * 16 + (lane & 15);
            const _Float16* bp = w1t + (long)n * DD + kk + khalf * 8;
            v8h blo = *(const v8h*)(bp);
            v8h bhi = *(const v8h*)(bp + 16);
            v16h bfrag;
#pragma unroll
            for (int i = 0; i < 8; ++i) { bfrag[i] = blo[i]; bfrag[8 + i] = bhi[i]; }
            acc[t] = __builtin_amdgcn_wmma_f32_16x16x32_f16(
                false, afrag, false, bfrag, (short)0, acc[t], false, false);
        }
    }

    // Per-lane column constants for the W2 reduction.
    float w2v[8], b1v[8];
#pragma unroll
    for (int t = 0; t < 8; ++t) {
        const int n = n0 + t * 16 + (lane & 15);
        w2v[t] = w2[n];
        b1v[t] = b1[n];
    }

    // p[v] = sum over this lane's N columns of relu(h)*W2.
    // C layout: VGPR v -> M=v (lanes 0-15) / M=v+8 (lanes 16-31), N=lane&15.
    float p[8];
#pragma unroll
    for (int v = 0; v < 8; ++v) {
        float s = 0.0f;
#pragma unroll
        for (int t = 0; t < 8; ++t) {
            float h = acc[t][v] + b1v[t];
            h = h > 0.0f ? h : 0.0f;
            s += h * w2v[t];
        }
        p[v] = s;
    }
    // Reduce across the 16 lanes of each half-wave (each half = one M group).
#pragma unroll
    for (int off = 8; off; off >>= 1)
#pragma unroll
        for (int v = 0; v < 8; ++v)
            p[v] += __shfl_xor(p[v], off, 16);

    if ((lane & 15) == 0) {
        const long mbase = t0 + (long)khalf * 8;
#pragma unroll
        for (int v = 0; v < 8; ++v)
            atomicAdd(&ys[mbase + v], p[v]);
    }
}

// ---------------------------------------------------------------------------
// Kernel 3: per-batch gate squash + max-lift + stable compaction scan.
// One 1024-thread block per batch row (4 sequential 1024-token rounds).
// ---------------------------------------------------------------------------
__global__ __launch_bounds__(1024)
void gate_scan(const float* __restrict__ ys,
               const unsigned char* __restrict__ pad,   // [B,S] bool bytes
               const float* __restrict__ b2,
               float* __restrict__ yf,                  // [B,S] final gate
               int* __restrict__ src_idx,               // [B,S] compaction map
               int* __restrict__ new_len)               // [B]
{
    const int b   = blockIdx.x;
    const int tid = threadIdx.x;
    const int lane = tid & 31, wid = tid >> 5;

    __shared__ float smax[32];
    __shared__ int   swave[32];
    __shared__ float s_adjust;
    __shared__ int   s_carry;

    const float bias = b2[0];
    float g[4];
    float lm = 0.0f;
#pragma unroll
    for (int r = 0; r < 4; ++r) {
        const int s = r * 1024 + tid;
        const float yraw = ys[b * SS + s] + bias;
        float gate = (1.0f + tanhf(10.0f * yraw)) * 0.5f;
        if (pad[b * SS + s]) gate = 0.0f;
        g[r] = gate;
        lm = fmaxf(lm, gate);
    }
#pragma unroll
    for (int off = 16; off; off >>= 1) lm = fmaxf(lm, __shfl_xor(lm, off, 32));
    if (lane == 0) smax[wid] = lm;
    __syncthreads();
    if (tid == 0) {
        float m = smax[0];
        for (int i = 1; i < 32; ++i) m = fmaxf(m, smax[i]);
        float adj = 1e-5f + 0.1f - m;
        s_adjust = adj > 0.0f ? adj : 0.0f;
        s_carry  = 0;
    }
    __syncthreads();
    const float adjust = s_adjust;

    for (int r = 0; r < 4; ++r) {
        const int s = r * 1024 + tid;
        const bool  p   = pad[b * SS + s] != 0;
        const float yfv = g[r] + adjust;
        yf[b * SS + s] = yfv;
        const bool keep = (yfv > 0.1f) && !p;

        const unsigned mask = (unsigned)__ballot(keep);
        const int excl = __popc(mask & ((1u << lane) - 1u));
        if (lane == 0) swave[wid] = __popc(mask);
        __syncthreads();
        if (tid == 0) {
            int run = s_carry;
            for (int i = 0; i < 32; ++i) { int t = swave[i]; swave[i] = run; run += t; }
            s_carry = run;
        }
        __syncthreads();
        const int pos = swave[wid] + excl;
        if (keep) src_idx[b * SS + pos] = s;
        __syncthreads();   // swave reused next round
    }
    if (tid == 0) new_len[b] = s_carry;
}

// ---------------------------------------------------------------------------
// Kernel 4: write outputs. One block per output row (b, j):
// j < new_len -> gated source row; else zero. Also emit v_pad as 0/1 floats.
// ---------------------------------------------------------------------------
__global__ __launch_bounds__(256)
void scatter_out(const float* __restrict__ x,
                 const float* __restrict__ yf,
                 const int* __restrict__ src_idx,
                 const int* __restrict__ new_len,
                 float* __restrict__ v_out,
                 float* __restrict__ pad_out)
{
    const int j = blockIdx.x & (SS - 1);
    const int b = blockIdx.x >> 12;
    const int len = new_len[b];
    const long obase = ((long)b * SS + j) * DD;

    if (j < len) {
        const int s = src_idx[b * SS + j];
        const float gv = yf[b * SS + s];
        const float* src = x + ((long)b * SS + s) * DD;
#pragma unroll
        for (int c = threadIdx.x * 4; c < DD; c += 1024) {
            v4f t = *(const v4f*)(src + c);
            t *= gv;
            *(v4f*)(v_out + obase + c) = t;
        }
    } else {
#pragma unroll
        for (int c = threadIdx.x * 4; c < DD; c += 1024) {
            *(v4f*)(v_out + obase + c) = (v4f)0.0f;
        }
    }
    if (threadIdx.x == 0) pad_out[b * SS + j] = (j >= len) ? 1.0f : 0.0f;
}

// ---------------------------------------------------------------------------
extern "C" void kernel_launch(void* const* d_in, const int* in_sizes, int n_in,
                              void* d_out, int out_size, void* d_ws, size_t ws_size,
                              hipStream_t stream) {
    const float*         x   = (const float*)d_in[0];
    const unsigned char* pad = (const unsigned char*)d_in[1];   // bool bytes
    const float*         W1  = (const float*)d_in[2];
    const float*         b1  = (const float*)d_in[3];
    const float*         W2  = (const float*)d_in[4];
    const float*         b2  = (const float*)d_in[5];
    float* out = (float*)d_out;

    char* ws = (char*)d_ws;
    _Float16* w1t = (_Float16*)(ws);                       // 1 MB
    float*    ysv = (float*)(ws + (1 << 20));              // 128 KB
    float*    yfv = (float*)(ws + (1 << 20) + (1 << 17));  // 128 KB
    int*      srci = (int*)(ws + (1 << 20) + (2 << 17));   // 128 KB
    int*      nlen = (int*)(ws + (1 << 20) + (3 << 17));   // 32 B

    // 1) pack W1 -> f16 transposed, zero ys accumulator
    pack_w1<<<(DH * DD) / 256, 256, 0, stream>>>(W1, w1t, ysv);
    // 2) fused gate MLP via WMMA: 8192 waves, 4 waves/block
    gate_mlp_wmma<<<(NTOK / 16) * 4 / 4, 128, 0, stream>>>(x, w1t, b1, W2, ysv);
    // 3) per-batch squash + lift + stable compaction scan
    gate_scan<<<BB, 1024, 0, stream>>>(ysv, pad, b2, yfv, srci, nlen);
    // 4) gated scatter + tail zero + v_pad
    scatter_out<<<NTOK, 256, 0, stream>>>(x, yfv, srci, nlen, out,
                                          out + (size_t)NTOK * DD);
}
